// MultiHeadSelfAttention_7026566496824
// MI455X (gfx1250) — matrix operand order repaired and checked by reading, not compiled, not run on hardware
//
#include <hip/hip_runtime.h>
#include <hip/hip_bf16.h>
#include <math.h>

typedef __attribute__((ext_vector_type(16))) _Float16 v16h;
typedef __attribute__((ext_vector_type(8)))  _Float16 v8h;
typedef __attribute__((ext_vector_type(8)))  float    v8f;

#define BATCH  2
#define SEQ    2048
#define DMODEL 1024
#define NHEADS 16
#define DKH    64
#define NQKV   3072
#define THETA  10000.0f

static __device__ __forceinline__ v16h cat8(v8h lo, v8h hi) {
  return __builtin_shufflevector(lo, hi, 0,1,2,3,4,5,6,7,8,9,10,11,12,13,14,15);
}
static __device__ __forceinline__ v8f wmma_f16(v16h a, v16h b, v8f c) {
  // v_wmma_f32_16x16x32_f16 : D = A(16x32) * B(32x16) + C(16x16 f32)
  return __builtin_amdgcn_wmma_f32_16x16x32_f16(false, a, false, b, (short)0, c,
                                                false, false);
}

// ---------------------------------------------------------------- convert ---
__global__ void cvt_f32_to_f16(const float* __restrict__ in,
                               _Float16* __restrict__ out, int n) {
  int i = blockIdx.x * blockDim.x + threadIdx.x;
  if (i < n) out[i] = (_Float16)in[i];
}

// ------------------------------------------------------------- RoPE table ---
// tab[s][fi] = (cos, sin) of tokpos[s] * THETA^(-2*fi/64), fi in [0,32)
__global__ void build_rope_tab(const int* __restrict__ tokpos,
                               float2* __restrict__ tab) {
  int i = blockIdx.x * blockDim.x + threadIdx.x;
  if (i >= SEQ * 32) return;
  int s  = i >> 5;
  int fi = i & 31;
  float inv = __powf(THETA, -(float)(2 * fi) * (1.0f / 64.0f));
  float ang = (float)tokpos[s] * inv;
  tab[i] = make_float2(__cosf(ang), __sinf(ang));
}

// --------------------------------------------------------- QKV GEMM + RoPE --
// C[m, o] = sum_i X[m, i] * W[o, i]   (M=4096, N=3072, K=1024)
#define BM  128
#define BN  128
#define BKK 32
#define LDT 40   // padded LDS row stride in halves (80B -> conflict-free b128)

__global__ __launch_bounds__(256)
void qkv_rope_gemm(const _Float16* __restrict__ X,
                   const _Float16* __restrict__ W,
                   _Float16* __restrict__ Qh, _Float16* __restrict__ Kh,
                   _Float16* __restrict__ Vh,
                   const float2* __restrict__ rope) {
  __shared__ _Float16 As[BM * LDT];
  __shared__ _Float16 Bs[BN * LDT];
  const int K    = DMODEL;
  const int m0   = blockIdx.y * BM;
  const int n0   = blockIdx.x * BN;
  const int tid  = threadIdx.x;
  const int lane = tid & 31;
  const int wid  = tid >> 5;
  const int wr   = wid & 3;   // 4 row-groups of 32
  const int wc   = wid >> 2;  // 2 col-groups of 64
  const int lrow = tid >> 1;            // 0..127
  const int lcol = (tid & 1) << 4;      // 0 or 16 halves

  v8f acc[2][4];
#pragma unroll
  for (int i = 0; i < 2; i++)
#pragma unroll
    for (int j = 0; j < 4; j++)
#pragma unroll
      for (int e = 0; e < 8; e++) acc[i][j][e] = 0.0f;

  for (int k0 = 0; k0 < K; k0 += BKK) {
    const v8h* ga = (const v8h*)(X + (size_t)(m0 + lrow) * K + k0 + lcol);
    v8h a0 = ga[0], a1 = ga[1];
    const v8h* gb = (const v8h*)(W + (size_t)(n0 + lrow) * K + k0 + lcol);
    v8h b0 = gb[0], b1 = gb[1];
    __syncthreads();
    *(v8h*)(As + lrow * LDT + lcol)     = a0;
    *(v8h*)(As + lrow * LDT + lcol + 8) = a1;
    *(v8h*)(Bs + lrow * LDT + lcol)     = b0;
    *(v8h*)(Bs + lrow * LDT + lcol + 8) = b1;
    __syncthreads();

    v16h af[2];
#pragma unroll
    for (int ti = 0; ti < 2; ti++) {
      int row = wr * 32 + ti * 16 + (lane & 15);
      int kb  = (lane >> 4) * 8;  // A layout: K 0-7/16-23 vs 8-15/24-31
      v8h lo = *(const v8h*)(As + row * LDT + kb);
      v8h hi = *(const v8h*)(As + row * LDT + kb + 16);
      af[ti] = cat8(lo, hi);
    }
    v16h bf[4];
#pragma unroll
    for (int tj = 0; tj < 4; tj++) {
      int col = wc * 64 + tj * 16 + (lane & 15);
      int kb  = (lane >> 4) * 8;  // B layout: K 0-7/16-23 vs 8-15/24-31
      v8h lo = *(const v8h*)(Bs + col * LDT + kb);
      v8h hi = *(const v8h*)(Bs + col * LDT + kb + 16);
      bf[tj] = cat8(lo, hi);
    }
#pragma unroll
    for (int ti = 0; ti < 2; ti++)
#pragma unroll
      for (int tj = 0; tj < 4; tj++)
        acc[ti][tj] = wmma_f16(af[ti], bf[tj], acc[ti][tj]);
  }

  // Epilogue: RoPE (table lookup) + scatter to [B,H,S,64] head layout.
#pragma unroll
  for (int ti = 0; ti < 2; ti++) {
    int mbase = m0 + wr * 32 + ti * 16;
#pragma unroll
    for (int tj = 0; tj < 4; tj++) {
      int o   = n0 + wc * 64 + tj * 16 + (lane & 15);
      int sec = o >> 10;        // 0=q 1=k 2=v (uniform across the wave)
      int oo  = o & 1023;
      int h   = oo >> 6;
      int dk  = oo & 63;
      int fi  = dk >> 1;
#pragma unroll
      for (int j = 0; j < 8; j++) {
        int M = j + ((lane >> 4) << 3);
        int m = mbase + M;
        int s = m & (SEQ - 1);
        int b = m >> 11;        // m / SEQ
        float v = acc[ti][tj][j];
        size_t idx = ((size_t)(b * NHEADS + h) * SEQ + s) * DKH + dk;
        if (sec < 2) {
          float  p  = __shfl_xor(v, 1);     // partner of the even/odd pair
          float2 cs = rope[s * 32 + fi];
          float  r  = (dk & 1) ? (p * cs.y + v * cs.x) : (v * cs.x - p * cs.y);
          ((sec == 0) ? Qh : Kh)[idx] = (_Float16)r;
        } else {
          Vh[idx] = (_Float16)v;
        }
      }
    }
  }
}

// ------------------------------------------------------------ flash attn ----
// Grid: (S/128, H, B). Block 256 = 8 waves; wave w owns 16 q-rows.
#define LDS_K 72   // padded row stride (144B, 16B aligned, conflict-free)

__global__ __launch_bounds__(256)
void attn_kernel(const _Float16* __restrict__ Qh, const _Float16* __restrict__ Kh,
                 const _Float16* __restrict__ Vh, _Float16* __restrict__ Yh) {
  __shared__ _Float16 Ks[64 * LDS_K];       // [key][dk]
  __shared__ _Float16 Vt[64 * LDS_K];       // [dk][key] (transposed)
  __shared__ _Float16 Ps[8 * 16 * LDS_K];   // per-wave P tiles [16][64]

  const int qblk = blockIdx.x;
  const int h    = blockIdx.y;
  const int b    = blockIdx.z;
  const int tid  = threadIdx.x;
  const int lane = tid & 31;
  const int wid  = tid >> 5;
  const size_t bh = (size_t)(b * NHEADS + h) * SEQ * DKH;
  const int qr0   = qblk * 128 + wid * 16;

  // Q fragments for this wave's 16 rows (2 K-steps over dk=64)
  v16h qf[2];
  {
    const _Float16* qp = Qh + bh + (size_t)(qr0 + (lane & 15)) * DKH;
#pragma unroll
    for (int st = 0; st < 2; st++) {
      int kb = st * 32 + (lane >> 4) * 8;
      v8h lo = *(const v8h*)(qp + kb);
      v8h hi = *(const v8h*)(qp + kb + 16);
      qf[st] = cat8(lo, hi);
    }
  }

  v8f oacc[4];
#pragma unroll
  for (int t = 0; t < 4; t++)
#pragma unroll
    for (int e = 0; e < 8; e++) oacc[t][e] = 0.0f;
  float m_i[8], l_i[8];
#pragma unroll
  for (int j = 0; j < 8; j++) { m_i[j] = -1e30f; l_i[j] = 0.0f; }

  // Staging indices (uniform participation; EXEC all-ones here)
  const int srow = tid >> 2;            // 0..63
  const int soff = (tid & 3) << 4;      // half offset: 0,16,32,48

  const int nkb = (qblk + 1) * 2;  // causal: 64-key blocks up to this q block
  for (int kb = 0; kb < nkb; kb++) {
    const int key0 = kb * 64;
    __syncthreads();
    {
      // --- K tile: CDNA5 async copy global -> LDS (ASYNCcnt-tracked) ---
      unsigned lds0 = (unsigned)(size_t)(Ks + srow * LDS_K + soff);
      unsigned long long g0 =
          (unsigned long long)(Kh + bh + (size_t)(key0 + srow) * DKH + soff);
      asm volatile("global_load_async_to_lds_b128 %0, %1, off"
                   :: "v"(lds0), "v"(g0) : "memory");
      unsigned lds1 = lds0 + 16;
      unsigned long long g1 = g0 + 16;
      asm volatile("global_load_async_to_lds_b128 %0, %1, off"
                   :: "v"(lds1), "v"(g1) : "memory");

      // --- V tile: load + transpose into Vt via register scatter ---
      const v8h* gv = (const v8h*)(Vh + bh + (size_t)(key0 + srow) * DKH + soff);
      v8h v0 = gv[0], v1 = gv[1];
#pragma unroll
      for (int i = 0; i < 8; i++) Vt[(soff + i) * LDS_K + srow]     = v0[i];
#pragma unroll
      for (int i = 0; i < 8; i++) Vt[(soff + 8 + i) * LDS_K + srow] = v1[i];

      asm volatile("s_wait_asynccnt 0x0" ::: "memory");
    }
    __syncthreads();

    // scores S = Q * K^T  (16 x 64, four 16-col tiles)
    v8f sc[4];
#pragma unroll
    for (int t = 0; t < 4; t++) {
#pragma unroll
      for (int e = 0; e < 8; e++) sc[t][e] = 0.0f;
#pragma unroll
      for (int st = 0; st < 2; st++) {
        int col = t * 16 + (lane & 15);
        int kb2 = st * 32 + (lane >> 4) * 8;
        v8h lo = *(const v8h*)(Ks + col * LDS_K + kb2);
        v8h hi = *(const v8h*)(Ks + col * LDS_K + kb2 + 16);
        sc[t] = wmma_f16(qf[st], cat8(lo, hi), sc[t]);
      }
    }

    // mask + online softmax (row = 16-lane half-group x accumulator slot j)
    const float scale = 0.125f;  // 1/sqrt(64)
#pragma unroll
    for (int j = 0; j < 8; j++) {
      int M    = j + ((lane >> 4) << 3);
      int qpos = qr0 + M;
      float mx = -1e30f;
#pragma unroll
      for (int t = 0; t < 4; t++) {
        int kpos = key0 + t * 16 + (lane & 15);
        float sv = sc[t][j] * scale;
        sv = (kpos <= qpos) ? sv : -1e30f;
        sc[t][j] = sv;
        mx = fmaxf(mx, sv);
      }
      for (int msk = 1; msk < 16; msk <<= 1) mx = fmaxf(mx, __shfl_xor(mx, msk));
      float mn = fmaxf(m_i[j], mx);
      float sum = 0.0f;
#pragma unroll
      for (int t = 0; t < 4; t++) {
        float p = __expf(sc[t][j] - mn);
        sc[t][j] = p;
        sum += p;
      }
      for (int msk = 1; msk < 16; msk <<= 1) sum += __shfl_xor(sum, msk);
      float corr = __expf(m_i[j] - mn);
      l_i[j] = l_i[j] * corr + sum;
      m_i[j] = mn;
#pragma unroll
      for (int t = 0; t < 4; t++) oacc[t][j] *= corr;
      // stash P row (f16) into this wave's private LDS tile
#pragma unroll
      for (int t = 0; t < 4; t++)
        Ps[(wid * 16 + M) * LDS_K + t * 16 + (lane & 15)] = (_Float16)sc[t][j];
    }

    // O += P * V   (A = P 16x64 from LDS, B = V^T columns from Vt)
#pragma unroll
    for (int st = 0; st < 2; st++) {
      int kbp = st * 32 + (lane >> 4) * 8;
      const _Float16* pp = Ps + (wid * 16 + (lane & 15)) * LDS_K;
      v8h plo = *(const v8h*)(pp + kbp);
      v8h phi = *(const v8h*)(pp + kbp + 16);
      v16h pf = cat8(plo, phi);
#pragma unroll
      for (int t = 0; t < 4; t++) {
        int col = t * 16 + (lane & 15);
        int kb2 = st * 32 + (lane >> 4) * 8;
        v8h blo = *(const v8h*)(Vt + col * LDS_K + kb2);
        v8h bhi = *(const v8h*)(Vt + col * LDS_K + kb2 + 16);
        oacc[t] = wmma_f16(pf, cat8(blo, bhi), oacc[t]);
      }
    }
  }

  // finalize: O / l, write Y in [B*S, 1024] f16 for the output GEMM
#pragma unroll
  for (int t = 0; t < 4; t++) {
#pragma unroll
    for (int j = 0; j < 8; j++) {
      int M = j + ((lane >> 4) << 3);
      int sr = qr0 + M;
      float y = oacc[t][j] / l_i[j];
      Yh[(size_t)(b * SEQ + sr) * DMODEL + h * DKH + t * 16 + (lane & 15)] =
          (_Float16)y;
    }
  }
}

// --------------------------------------------------------- output GEMM ------
// C[m, n] = sum_i A[m, i] * W[n, i], f32 output.
__global__ __launch_bounds__(256)
void out_gemm(const _Float16* __restrict__ A, const _Float16* __restrict__ W,
              float* __restrict__ C, int N, int K) {
  __shared__ _Float16 As[BM * LDT];
  __shared__ _Float16 Bs[BN * LDT];
  const int m0   = blockIdx.y * BM;
  const int n0   = blockIdx.x * BN;
  const int tid  = threadIdx.x;
  const int lane = tid & 31;
  const int wid  = tid >> 5;
  const int wr   = wid & 3;
  const int wc   = wid >> 2;
  const int lrow = tid >> 1;
  const int lcol = (tid & 1) << 4;

  v8f acc[2][4];
#pragma unroll
  for (int i = 0; i < 2; i++)
#pragma unroll
    for (int j = 0; j < 4; j++)
#pragma unroll
      for (int e = 0; e < 8; e++) acc[i][j][e] = 0.0f;

  for (int k0 = 0; k0 < K; k0 += BKK) {
    const v8h* ga = (const v8h*)(A + (size_t)(m0 + lrow) * K + k0 + lcol);
    v8h a0 = ga[0], a1 = ga[1];
    const v8h* gb = (const v8h*)(W + (size_t)(n0 + lrow) * K + k0 + lcol);
    v8h b0 = gb[0], b1 = gb[1];
    __syncthreads();
    *(v8h*)(As + lrow * LDT + lcol)     = a0;
    *(v8h*)(As + lrow * LDT + lcol + 8) = a1;
    *(v8h*)(Bs + lrow * LDT + lcol)     = b0;
    *(v8h*)(Bs + lrow * LDT + lcol + 8) = b1;
    __syncthreads();

    v16h af[2];
#pragma unroll
    for (int ti = 0; ti < 2; ti++) {
      int row = wr * 32 + ti * 16 + (lane & 15);
      int kb  = (lane >> 4) * 8;
      v8h lo = *(const v8h*)(As + row * LDT + kb);
      v8h hi = *(const v8h*)(As + row * LDT + kb + 16);
      af[ti] = cat8(lo, hi);
    }
    v16h bf[4];
#pragma unroll
    for (int tj = 0; tj < 4; tj++) {
      int col = wc * 64 + tj * 16 + (lane & 15);
      int kb  = (lane >> 4) * 8;
      v8h lo = *(const v8h*)(Bs + col * LDT + kb);
      v8h hi = *(const v8h*)(Bs + col * LDT + kb + 16);
      bf[tj] = cat8(lo, hi);
    }
#pragma unroll
    for (int ti = 0; ti < 2; ti++)
#pragma unroll
      for (int tj = 0; tj < 4; tj++)
        acc[ti][tj] = wmma_f16(af[ti], bf[tj], acc[ti][tj]);
  }

#pragma unroll
  for (int ti = 0; ti < 2; ti++) {
#pragma unroll
    for (int tj = 0; tj < 4; tj++) {
      int n = n0 + wc * 64 + tj * 16 + (lane & 15);
#pragma unroll
      for (int j = 0; j < 8; j++) {
        int m = m0 + wr * 32 + ti * 16 + j + ((lane >> 4) << 3);
        C[(size_t)m * N + n] = acc[ti][tj][j];
      }
    }
  }
}

// ---------------------------------------------------------------- launch ----
extern "C" void kernel_launch(void* const* d_in, const int* in_sizes, int n_in,
                              void* d_out, int out_size, void* d_ws, size_t ws_size,
                              hipStream_t stream) {
  const float* x     = (const float*)d_in[0];   // [2,2048,1024]
  const float* w_qkv = (const float*)d_in[1];   // [3072,1024]
  const float* w_o   = (const float*)d_in[2];   // [1024,1024]
  const int*   tpos  = (const int*)d_in[3];     // [2048]
  float* out = (float*)d_out;                   // [2,2048,1024]

  char* p = (char*)d_ws;
  _Float16* xh  = (_Float16*)p; p += (size_t)BATCH * SEQ * DMODEL * 2;
  _Float16* wh  = (_Float16*)p; p += (size_t)NQKV * DMODEL * 2;
  _Float16* woh = (_Float16*)p; p += (size_t)DMODEL * DMODEL * 2;
  _Float16* Qh  = (_Float16*)p; p += (size_t)BATCH * NHEADS * SEQ * DKH * 2;
  _Float16* Kh  = (_Float16*)p; p += (size_t)BATCH * NHEADS * SEQ * DKH * 2;
  _Float16* Vh  = (_Float16*)p; p += (size_t)BATCH * NHEADS * SEQ * DKH * 2;
  _Float16* Yh  = (_Float16*)p; p += (size_t)BATCH * SEQ * DMODEL * 2;
  float2*   rt  = (float2*)p;   p += (size_t)SEQ * 32 * sizeof(float2);

  {
    int n = BATCH * SEQ * DMODEL;
    cvt_f32_to_f16<<<(n + 255) / 256, 256, 0, stream>>>(x, xh, n);
  }
  {
    int n = NQKV * DMODEL;
    cvt_f32_to_f16<<<(n + 255) / 256, 256, 0, stream>>>(w_qkv, wh, n);
  }
  {
    int n = DMODEL * DMODEL;
    cvt_f32_to_f16<<<(n + 255) / 256, 256, 0, stream>>>(w_o, woh, n);
  }
  {
    int n = SEQ * 32;
    build_rope_tab<<<(n + 255) / 256, 256, 0, stream>>>(tpos, rt);
  }

  dim3 g1(NQKV / BN, (BATCH * SEQ) / BM);
  qkv_rope_gemm<<<g1, 256, 0, stream>>>(xh, wh, Qh, Kh, Vh, rt);

  dim3 g2(SEQ / 128, NHEADS, BATCH);
  attn_kernel<<<g2, 256, 0, stream>>>(Qh, Kh, Vh, Yh);

  dim3 g3(DMODEL / BN, (BATCH * SEQ) / BM);
  out_gemm<<<g3, 256, 0, stream>>>(Yh, woh, out, DMODEL, DMODEL);
}
